// RBM_CB_32246614459250
// MI455X (gfx1250) — matrix operand order repaired and checked by reading, not compiled, not run on hardware
//
#include <hip/hip_runtime.h>

// Problem constants (match reference).
#define B_SZ 16384
#define NV   200
#define NC   10
#define NCP  16    // categories padded to WMMA N tile
#define NH   128
#define NVC  25    // v-chunk per block in sample_V (gridDim.y = NV/NVC = 8)

typedef __attribute__((ext_vector_type(16))) _Float16 v16h;
typedef __attribute__((ext_vector_type(8)))  float    v8f;
typedef __attribute__((ext_vector_type(4)))  unsigned u32x4;
typedef __attribute__((ext_vector_type(8)))  unsigned u32x8;

// ---------------- stateless RNG ----------------
__device__ __forceinline__ unsigned hmix(unsigned x) {
  x ^= x >> 16; x *= 0x7feb352du;
  x ^= x >> 15; x *= 0x846ca68bu;
  x ^= x >> 16; return x;
}
// one hash -> two uniforms in (0,1), 16-bit granularity
__device__ __forceinline__ unsigned hash3(unsigned seed, unsigned a, unsigned b) {
  return hmix(seed * 0x9E3779B1u ^ a * 0x85EBCA77u ^ b * 0xC2B2AE3Du ^ 0x27220A95u);
}
__device__ __forceinline__ float u16lo(unsigned x) {
  return (float)(x & 0xFFFFu) * (1.0f / 65536.0f) + (0.5f / 65536.0f);
}
__device__ __forceinline__ float u16hi(unsigned x) {
  return (float)(x >> 16) * (1.0f / 65536.0f) + (0.5f / 65536.0f);
}

// ---------------- TDM: one-instruction 4KB tile DMA (global -> LDS) ----------------
// D# group0: count=1 | lds_addr | global_addr | type=2 ("image").
// D# group1: data_size=4B, 1-D tile: tensor_dim0 = tile_dim0 = 1024 dwords.
__device__ __forceinline__ void tdm_load_tile(unsigned lds_off, const void* gaddr) {
  unsigned long long ga = (unsigned long long)gaddr;
  u32x4 g0;
  g0.x = 1u;                                        // count=1 (valid), user mode
  g0.y = lds_off;                                   // lds_addr (bytes)
  g0.z = (unsigned)ga;                              // global_addr[31:0]
  g0.w = ((unsigned)(ga >> 32) & 0x01FFFFFFu) | (2u << 30);  // addr[56:32] | type=2
  u32x8 g1;
  g1[0] = 2u << 16;        // workgroup_mask=0, data_size=2 (4 bytes), no flags
  g1[1] = 1024u << 16;     // tensor_dim0[15:0] (=1024) in bits 63:48
  g1[2] = 1u << 16;        // tensor_dim0 hi = 0 | tensor_dim1 lo = 1
  g1[3] = 1024u << 16;     // tensor_dim1 hi = 0 | tile_dim0 = 1024
  g1[4] = 0u;              // tile_dim1 = 0 (unused), tile_dim2 = 0
  g1[5] = 1024u;           // tensor_dim0_stride[31:0]
  g1[6] = 0u;              // stride hi / dim1_stride lo
  g1[7] = 0u;
  asm volatile("tensor_load_to_lds %0, %1" :: "s"(g0), "s"(g1) : "memory");
}

// ---------------- prep: W -> f16 WMMA A-layout (Wv^T tiles), Bb -> padded ----------------
__global__ void prep_kernel(const float* __restrict__ W, const float* __restrict__ Bb,
                            _Float16* __restrict__ WA, float* __restrict__ Bbp) {
  unsigned e = blockIdx.x * blockDim.x + threadIdx.x;   // 0 .. 409599
  unsigned j = e & 15u, l = (e >> 4) & 31u, t = (e >> 9) & 3u, v = e >> 11;
  if (v < NV) {
    unsigned m = l & 15u, hs = l >> 4;
    unsigned k = j + 16u * hs;
    unsigned hidx = t * 32u + k;
    float w = (m < NC) ? W[((size_t)(v * NC + m)) * NH + hidx] : 0.0f;
    WA[e] = (_Float16)w;
  }
  if (e < NV * NCP) {
    unsigned v2 = e / NCP, n2 = e % NCP;
    Bbp[e] = (n2 < NC) ? Bb[v2 * NC + n2] : -1e30f;
  }
}

// ---------------- extract one-hot -> index ----------------
__global__ void extract_idx_kernel(const float* __restrict__ V0,
                                   unsigned char* __restrict__ idx) {
  unsigned t = blockIdx.x * blockDim.x + threadIdx.x;  // t = b*NV + v (exact grid)
  const float* p = V0 + (size_t)t * NC;
  float best = p[0]; int bi = 0;
#pragma unroll
  for (int k = 1; k < NC; ++k) { float x = p[k]; if (x > best) { best = x; bi = k; } }
  idx[t] = (unsigned char)bi;
}

// ---------------- sample_h: gather-sum (one-hot GEMM collapses to gathers) ----------------
__global__ void __launch_bounds__(256) sample_h_kernel(
    const unsigned char* __restrict__ idx, const float* __restrict__ W,
    const float* __restrict__ c, unsigned seed,
    float* __restrict__ h_f32, _Float16* __restrict__ h_f16) {
  unsigned b    = (blockIdx.x * blockDim.x + threadIdx.x) >> 5;
  unsigned lane = threadIdx.x & 31u;
  const unsigned char* ib = idx + (size_t)b * NV;
  float4 acc = {0.f, 0.f, 0.f, 0.f};
  for (int v = 0; v < NV; v += 4) {
    unsigned iw = *(const unsigned*)(ib + v);          // 4 packed indices
#pragma unroll
    for (int k = 0; k < 4; ++k) {
      unsigned cv = (iw >> (8 * k)) & 255u;
      float4 w4 = ((const float4*)(W + ((size_t)((v + k) * NC + cv)) * NH))[lane];
      acc.x += w4.x; acc.y += w4.y; acc.z += w4.z; acc.w += w4.w;
    }
  }
  float4 cb = ((const float4*)c)[lane];
  float q[4] = {acc.x + cb.x, acc.y + cb.y, acc.z + cb.z, acc.w + cb.w};
  unsigned x0 = hash3(seed, b * 2u,      lane);
  unsigned x1 = hash3(seed, b * 2u + 1u, lane);
  float u[4] = {u16lo(x0), u16hi(x0), u16lo(x1), u16hi(x1)};
  union { _Float16 h[4]; uint2 u2; } hp;
  float hv[4];
#pragma unroll
  for (int k = 0; k < 4; ++k) {
    float p = 1.0f / (1.0f + __expf(-q[k]));
    hv[k] = (u[k] < p) ? 1.0f : 0.0f;
    hp.h[k] = (_Float16)hv[k];
  }
  if (h_f32) {
    float4 o = {hv[0], hv[1], hv[2], hv[3]};
    ((float4*)(h_f32 + (size_t)b * NH))[lane] = o;
  }
  *(uint2*)(h_f16 + (size_t)b * NH + lane * 4u) = hp.u2;
}

// ---------------- sample_V: Y^T[16c x 16b] = Wv^T x h^T per visible unit ------
// A (Wv^T) double-buffered in LDS via one TDM DMA per tile; B (h^T) register-resident.
// D layout: lane = batch col, VGPR r = category -> in-register max + ONE shfl_xor(16).
__global__ void __launch_bounds__(256) sample_V_kernel(
    const _Float16* __restrict__ h_f16, const _Float16* __restrict__ WA,
    const float* __restrict__ Bbp, unsigned seed,
    unsigned char* __restrict__ idx_out) {
  __shared__ v16h  sB[2][128];                     // two 4KB tiles
  __shared__ float sBias[NVC * NCP];               // 1.6KB bias slab for this v-chunk
  const unsigned tid  = threadIdx.x;
  const unsigned lane = tid & 31u, wv = tid >> 5;
  const unsigned n = lane & 15u, hs = lane >> 4;
  const unsigned b = blockIdx.x * 128u + wv * 16u + n;
  const int v0 = (int)blockIdx.y * NVC, v1 = v0 + NVC;

  // B operand: h^T for this wave's 16 batch rows (contiguous per lane).
  v16h bmat[4];
  {
    const v16h* hb = (const v16h*)(h_f16 + (size_t)b * NH + hs * 16u);
#pragma unroll
    for (int t = 0; t < 4; ++t) bmat[t] = hb[t * 2];
  }
  // Stage bias slab.
  for (unsigned i = tid; i < NVC * NCP; i += 256u) sBias[i] = Bbp[(unsigned)v0 * NCP + i];

  const unsigned ldsbase = (unsigned)(size_t)(&sB[0][0]);  // LDS offset (low 32 bits)
  if (wv == 0) tdm_load_tile(ldsbase, (const char*)WA + (size_t)v0 * 4096);

  for (int v = v0; v < v1; ++v) {
    if (wv == 0) __builtin_amdgcn_s_wait_tensorcnt(0);
    __syncthreads();
    if (wv == 0 && v + 1 < v1)
      tdm_load_tile(ldsbase + ((((unsigned)(v + 1 - v0)) & 1u) << 12),
                    (const char*)WA + (size_t)(v + 1) * 4096);

    const v16h* at = sB[(v - v0) & 1];
    v8f acc = {};
#pragma unroll
    for (int t = 0; t < 4; ++t)
      acc = __builtin_amdgcn_wmma_f32_16x16x32_f16(
          false, at[t * 32 + lane], false, bmat[t], (short)0, acc, false, false);

    // Gumbel-argmax over this lane's categories (pairs share one hash).
    const unsigned bvv = b * NV + (unsigned)v;
    const float* bias = sBias + (unsigned)(v - v0) * NCP;
    unsigned key = 0u;
#pragma unroll
    for (int r = 0; r < 8; r += 2) {
      unsigned cat = (unsigned)r + 8u * hs;
      if (cat < NC) {                                        // cat even -> cat+1 also real
        unsigned x = hash3(seed, bvv, cat);
        float ta = -__builtin_amdgcn_logf(u16lo(x)) * 0.69314718f;
        float tb = -__builtin_amdgcn_logf(u16hi(x)) * 0.69314718f;
        float ga = -__builtin_amdgcn_logf(ta) * 0.69314718f;
        float gb = -__builtin_amdgcn_logf(tb) * 0.69314718f;
        float ya = acc[r]     + bias[cat]     + ga;
        float yb = acc[r + 1] + bias[cat + 1] + gb;
        unsigned ua = __float_as_uint(ya);
        ua = (ua & 0x80000000u) ? ~ua : (ua | 0x80000000u);
        unsigned ka = (ua & 0xFFFFFFF0u) | cat;
        unsigned ub = __float_as_uint(yb);
        ub = (ub & 0x80000000u) ? ~ub : (ub | 0x80000000u);
        unsigned kb = (ub & 0xFFFFFFF0u) | (cat + 1u);
        unsigned km = (ka > kb) ? ka : kb;
        key = (km > key) ? km : key;
      }
    }
    unsigned o = __shfl_xor(key, 16, 32);                    // merge the two halves
    key = (o > key) ? o : key;
    if (hs == 0) idx_out[(size_t)b * NV + v] = (unsigned char)(key & 15u);
  }
}

// ---------------- one-hot materialization (streaming 131MB write, b128 stores) ----------------
__global__ void onehot_kernel(const unsigned char* __restrict__ idx,
                              float* __restrict__ V) {
  unsigned t = blockIdx.x * blockDim.x + threadIdx.x;   // pair of cells, exact grid
  unsigned pb = *(const unsigned short*)(idx + 2u * t);
  unsigned biA = pb & 255u, biB = (pb >> 8) + 10u;      // second cell offset by 10
  float4* p = (float4*)(V + (size_t)t * 20u);
#pragma unroll
  for (int q = 0; q < 5; ++q) {
    float4 o;
    o.x = ((unsigned)(4 * q)     == biA || (unsigned)(4 * q)     == biB) ? 1.0f : 0.0f;
    o.y = ((unsigned)(4 * q + 1) == biA || (unsigned)(4 * q + 1) == biB) ? 1.0f : 0.0f;
    o.z = ((unsigned)(4 * q + 2) == biA || (unsigned)(4 * q + 2) == biB) ? 1.0f : 0.0f;
    o.w = ((unsigned)(4 * q + 3) == biA || (unsigned)(4 * q + 3) == biB) ? 1.0f : 0.0f;
    p[q] = o;
  }
}

// ---------------- launch ----------------
extern "C" void kernel_launch(void* const* d_in, const int* in_sizes, int n_in,
                              void* d_out, int out_size, void* d_ws, size_t ws_size,
                              hipStream_t stream) {
  (void)in_sizes; (void)n_in; (void)out_size; (void)ws_size;
  const float* V0 = (const float*)d_in[0];
  const float* W  = (const float*)d_in[1];
  const float* Bb = (const float*)d_in[2];
  const float* c  = (const float*)d_in[3];

  float* V_out  = (float*)d_out;                                  // [B,NV,NC]
  float* h_out  = V_out + (size_t)B_SZ * NV * NC;                 // [B,NH]
  float* h0_out = h_out + (size_t)B_SZ * NH;                      // [B,NH]

  char* ws = (char*)d_ws;
  _Float16*      WAf  = (_Float16*)ws;                            //   819,200 B
  float*         Bbp  = (float*)(ws + 819200);                    //    12,800 B
  unsigned char* idxb = (unsigned char*)(ws + 832000);            // 3,276,800 B
  _Float16*      hf16 = (_Float16*)(ws + 4108800);                // 4,194,304 B

  prep_kernel<<<1600, 256, 0, stream>>>(W, Bb, WAf, Bbp);
  extract_idx_kernel<<<(B_SZ * NV) / 256, 256, 0, stream>>>(V0, idxb);

  // h_given_V_0 (also initial h of the chain)
  sample_h_kernel<<<(B_SZ * 32) / 256, 256, 0, stream>>>(idxb, W, c, 0u, h0_out, hf16);

  // Gibbs iteration 0
  sample_V_kernel<<<dim3(B_SZ / 128, NV / NVC), 256, 0, stream>>>(hf16, WAf, Bbp, 1u, idxb);
  sample_h_kernel<<<(B_SZ * 32) / 256, 256, 0, stream>>>(idxb, W, c, 2u, nullptr, hf16);

  // Gibbs iteration 1 (final)
  sample_V_kernel<<<dim3(B_SZ / 128, NV / NVC), 256, 0, stream>>>(hf16, WAf, Bbp, 3u, idxb);
  sample_h_kernel<<<(B_SZ * 32) / 256, 256, 0, stream>>>(idxb, W, c, 4u, h_out, hf16);

  // Materialize one-hot V_given_h from final indices
  onehot_kernel<<<(B_SZ * NV) / 512, 256, 0, stream>>>(idxb, V_out);
}
